// PredictiveAttentionRouter_81844896792946
// MI455X (gfx1250) — compile-verified, matrix operand order checked by reading, not run on hardware
//
#include <hip/hip_runtime.h>
#include <hip/hip_bf16.h>
#include <math.h>

// ---------------------------------------------------------------------------
// PredictiveAttentionRouter for MI455X (gfx1250, wave32, WMMA 16x16x32 f16)
//
// Stage 1: memproj  : mem[m,o] = memory[m,:]·Wm[o,:] + bm[o]  -> f16 twice:
//                     memF16[h][m][d]  (B operand for scores GEMM, [N][K])
//                     memT16[h][d][m]  (B operand for output GEMM, [N][K])
// Stage 2: gate     : gate[b,t,h] = sigmoid(h[b,t,:]·Wg[h,:] + bg[h])
// Stage 3: attn     : per (b, head, 64-token tile):
//                     - 64 KB projected-memory L2->LDS fill via
//                       GLOBAL_LOAD_ASYNC_TO_LDS_B128 (ASYNCcnt), overlapped
//                       with the Q-tile f32->f16 conversion
//                     - S = Q @ Mem^T (WMMA), scale by gate/8, softmax over 256
//                       with in-register max/sum via shfl_xor (lane-half safe)
//                     - unnormalized exp (f16) -> LDS, O = P @ Mem (WMMA)
//                     - divide by row sum, store f32
// ---------------------------------------------------------------------------

#define D_MODEL 1024
#define N_HEADS 16
#define HEAD_DIM 64
#define N_MEM 256
#define T_TOK 4096
#define BATCH 4
#define T_TILE 64            // tokens per workgroup (4 waves x 16 rows)

typedef _Float16 v16h __attribute__((ext_vector_type(16)));
typedef _Float16 v8h  __attribute__((ext_vector_type(8)));
typedef _Float16 v4h  __attribute__((ext_vector_type(4)));
typedef float    v8f  __attribute__((ext_vector_type(8)));

// ---------------------------------------------------------------------------
// Stage 1: memory projection (tiny: 0.5 GFLOP). Scalar f32 dot per output.
// ---------------------------------------------------------------------------
__global__ __launch_bounds__(256)
void PAR_memproj_kernel(const float* __restrict__ memory,
                        const float* __restrict__ Wm,
                        const float* __restrict__ bm,
                        _Float16* __restrict__ memF16,
                        _Float16* __restrict__ memT16) {
    int idx = blockIdx.x * 256 + threadIdx.x;      // 0 .. 256*1024-1
    int m = idx >> 10;                             // 0..255
    int o = idx & 1023;                            // 0..1023
    const float4* mr = (const float4*)(memory + (size_t)m * D_MODEL);
    const float4* wr = (const float4*)(Wm + (size_t)o * D_MODEL);
    float acc = bm[o];
#pragma unroll 4
    for (int i = 0; i < D_MODEL / 4; ++i) {
        float4 a = mr[i], w = wr[i];
        acc += a.x * w.x + a.y * w.y + a.z * w.z + a.w * w.w;
    }
    int head = o >> 6;          // o / HEAD_DIM
    int dd   = o & 63;          // o % HEAD_DIM
    memF16[((size_t)head * N_MEM + m) * HEAD_DIM + dd] = (_Float16)acc;
    memT16[((size_t)head * HEAD_DIM + dd) * N_MEM + m] = (_Float16)acc;
}

// ---------------------------------------------------------------------------
// Stage 2: sigmoid gate. One thread per (token, head). Streams h once.
// ---------------------------------------------------------------------------
__global__ __launch_bounds__(256)
void PAR_gate_kernel(const float* __restrict__ h,
                     const float* __restrict__ Wg,
                     const float* __restrict__ bg,
                     float* __restrict__ gate) {
    int idx = blockIdx.x * 256 + threadIdx.x;      // 0 .. 16384*16-1
    int tok = idx >> 4;
    int hh  = idx & 15;
    const float4* hr = (const float4*)(h + (size_t)tok * D_MODEL);
    const float4* wr = (const float4*)(Wg + (size_t)hh * D_MODEL);
    float acc = bg[hh];
#pragma unroll 4
    for (int i = 0; i < D_MODEL / 4; ++i) {
        float4 a = hr[i], w = wr[i];
        acc += a.x * w.x + a.y * w.y + a.z * w.z + a.w * w.w;
    }
    gate[idx] = 1.0f / (1.0f + __expf(-acc));
}

// ---------------------------------------------------------------------------
// CDNA5 async global->LDS copy (ASYNCcnt-tracked, no VGPR round trip).
// lds_off is the raw byte offset inside the wave's LDS allocation (dynamic
// shared memory starts at offset 0 here since the kernel has no static LDS).
// ---------------------------------------------------------------------------
__device__ __forceinline__
void PAR_async_b128(unsigned int lds_off, const void* gaddr) {
    asm volatile("global_load_async_to_lds_b128 %0, %1, off"
                 :: "v"(lds_off), "v"(gaddr) : "memory");
}
__device__ __forceinline__ void PAR_wait_async() {
    asm volatile("s_wait_asynccnt 0x0" ::: "memory");
}

// ---------------------------------------------------------------------------
// WMMA fragment loader. Source array is [Row][Col] row-major f16 in LDS with
// `stride` halves per row. 16-bit 16x32 A layout (and matching B layout on an
// [N][K]-major array):
//   lanes 0-15 : row = lane,    K = {k0+0..7,  k0+16..23}
//   lanes 16-31: row = lane-16, K = {k0+8..15, k0+24..31}
// Each half is one contiguous 16-byte ds_load_b128.
// ---------------------------------------------------------------------------
__device__ __forceinline__
v16h PAR_load_frag(const _Float16* base, int row0, int stride, int k0, int lane) {
    int r = row0 + (lane & 15);
    int kb = k0 + ((lane >> 4) << 3);              // +0 or +8
    const _Float16* p = base + (size_t)r * stride + kb;
    union { v16h v; v8h h[2]; } u;
    u.h[0] = *(const v8h*)(p);                     // K = kb .. kb+7
    u.h[1] = *(const v8h*)(p + 16);                // K = kb+16 .. kb+23
    return u.v;
}

// ---------------------------------------------------------------------------
// Stage 3: fused gated attention. Block = 128 threads (4 waves), each wave
// owns a 16-token row tile. 64 v_wmma_f32_16x16x32_f16 per wave.
// ---------------------------------------------------------------------------
__global__ __launch_bounds__(128)
void PAR_attn_kernel(const float* __restrict__ h,
                     const _Float16* __restrict__ memF16,
                     const _Float16* __restrict__ memT16,
                     const float* __restrict__ gate,
                     float* __restrict__ out) {
    const int tile = blockIdx.x;     // 0..63
    const int head = blockIdx.y;     // 0..15
    const int b    = blockIdx.z;     // 0..3
    const int t0   = tile * T_TILE;
    const int tid  = threadIdx.x;

    extern __shared__ char smem[];
    _Float16* sMem  = (_Float16*)(smem);                 // [256][64]  32 KB (LDS off 0)
    _Float16* sMemT = (_Float16*)(smem + 32768);         // [64][256]  32 KB (LDS off 32768)
    _Float16* sQ    = (_Float16*)(smem + 65536);         // [64][64]    8 KB
    _Float16* sP    = (_Float16*)(smem + 73728);         // 4x[16][256] 32 KB
    float*    sGate = (float*)(smem + 106496);           // [64]       256 B

    // ---- async L2->LDS fill of projected memory (both layouts) -------------
    {
        const uint4* g0 = (const uint4*)(memF16 + (size_t)head * N_MEM * HEAD_DIM);
        const uint4* g1 = (const uint4*)(memT16 + (size_t)head * HEAD_DIM * N_MEM);
#pragma unroll 4
        for (int i = tid; i < 2048; i += 128)
            PAR_async_b128((unsigned int)(i * 16), (const void*)(g0 + i));
#pragma unroll 4
        for (int i = tid; i < 2048; i += 128)
            PAR_async_b128((unsigned int)(32768 + i * 16), (const void*)(g1 + i));
    }

    // ---- Q tile (overlaps with the async fill): f32 -> f16 LDS -------------
    for (int i = tid; i < 1024; i += 128) {
        int r  = i >> 4;
        int c4 = (i & 15) << 2;
        float4 f = *(const float4*)(h + ((size_t)(b * T_TOK + t0 + r)) * D_MODEL
                                      + head * HEAD_DIM + c4);
        v4h q;
        q[0] = (_Float16)f.x; q[1] = (_Float16)f.y;
        q[2] = (_Float16)f.z; q[3] = (_Float16)f.w;
        *(v4h*)(sQ + r * HEAD_DIM + c4) = q;
    }
    if (tid < T_TILE)
        sGate[tid] = gate[((size_t)(b * T_TOK + t0 + tid)) * N_HEADS + head];

    PAR_wait_async();     // this wave's async fills done
    __syncthreads();      // all waves' fills + Q/gate visible

    const int wave = tid >> 5;
    const int lane = tid & 31;
    const int r0   = wave << 4;                 // wave's row-tile base
    const int hsel = (lane >> 4) << 3;          // 0 for lanes 0-15, 8 for 16-31
    const int col0 = lane & 15;

    // ---- GEMM1: scores S[16][256] = Q[16][64] @ Mem^T ----------------------
    v16h a0 = PAR_load_frag(sQ, r0, HEAD_DIM, 0,  lane);
    v16h a1 = PAR_load_frag(sQ, r0, HEAD_DIM, 32, lane);

    v8f acc[16];
#pragma unroll
    for (int j = 0; j < 16; ++j) {
        v16h b0 = PAR_load_frag(sMem, j * 16, HEAD_DIM, 0,  lane);
        v16h b1 = PAR_load_frag(sMem, j * 16, HEAD_DIM, 32, lane);
        v8f c = {};
        c = __builtin_amdgcn_wmma_f32_16x16x32_f16(false, a0, false, b0,
                                                   (short)0, c, false, false);
        c = __builtin_amdgcn_wmma_f32_16x16x32_f16(false, a1, false, b1,
                                                   (short)0, c, false, false);
        acc[j] = c;
    }

    // ---- gate scale + softmax stats (D layout: vgpr k -> row r0+k+hsel) ----
    float gv[8], mx[8], sum[8];
#pragma unroll
    for (int k = 0; k < 8; ++k) {
        gv[k]  = sGate[r0 + hsel + k] * 0.125f;   // gate / sqrt(HEAD_DIM)
        mx[k]  = -1e30f;
        sum[k] = 0.0f;
    }

#pragma unroll
    for (int j = 0; j < 16; ++j)
#pragma unroll
        for (int k = 0; k < 8; ++k) {
            float v = acc[j][k] * gv[k];
            acc[j][k] = v;
            mx[k] = fmaxf(mx[k], v);
        }
    // row-max across the 16 lanes of each half (xor <= 8 stays in-half)
#pragma unroll
    for (int m = 1; m <= 8; m <<= 1)
#pragma unroll
        for (int k = 0; k < 8; ++k)
            mx[k] = fmaxf(mx[k], __shfl_xor(mx[k], m, 32));

    // unnormalized exp -> f16 LDS in [row][m] layout (A operand of GEMM2)
    _Float16* pw = sP + (size_t)wave * 16 * N_MEM;
#pragma unroll
    for (int j = 0; j < 16; ++j)
#pragma unroll
        for (int k = 0; k < 8; ++k) {
            float e = __expf(acc[j][k] - mx[k]);
            sum[k] += e;
            pw[(size_t)(hsel + k) * N_MEM + j * 16 + col0] = (_Float16)e;
        }
#pragma unroll
    for (int m = 1; m <= 8; m <<= 1)
#pragma unroll
        for (int k = 0; k < 8; ++k)
            sum[k] += __shfl_xor(sum[k], m, 32);

    __syncthreads();   // make P stores visible before fragment loads

    // ---- GEMM2: O[16][64] = P[16][256] @ Mem[256][64] ----------------------
    v8f acc2[4] = {};
#pragma unroll
    for (int kc = 0; kc < 8; ++kc) {
        v16h pa = PAR_load_frag(pw, 0, N_MEM, kc * 32, lane);
#pragma unroll
        for (int jd = 0; jd < 4; ++jd) {
            v16h bf = PAR_load_frag(sMemT, jd * 16, N_MEM, kc * 32, lane);
            acc2[jd] = __builtin_amdgcn_wmma_f32_16x16x32_f16(false, pa, false, bf,
                                                              (short)0, acc2[jd],
                                                              false, false);
        }
    }

    // ---- normalize by row sum and store f32 --------------------------------
    float inv[8];
#pragma unroll
    for (int k = 0; k < 8; ++k) inv[k] = 1.0f / sum[k];

#pragma unroll
    for (int jd = 0; jd < 4; ++jd)
#pragma unroll
        for (int k = 0; k < 8; ++k) {
            int row = r0 + hsel + k;
            out[((size_t)(b * T_TOK + t0 + row)) * D_MODEL
                + head * HEAD_DIM + jd * 16 + col0] = acc2[jd][k] * inv[k];
        }
}

// ---------------------------------------------------------------------------
extern "C" void kernel_launch(void* const* d_in, const int* in_sizes, int n_in,
                              void* d_out, int out_size, void* d_ws, size_t ws_size,
                              hipStream_t stream) {
    const float* h      = (const float*)d_in[0];
    const float* memory = (const float*)d_in[1];
    const float* Wm     = (const float*)d_in[2];
    const float* bm     = (const float*)d_in[3];
    const float* Wg     = (const float*)d_in[4];
    const float* bg     = (const float*)d_in[5];
    float* out = (float*)d_out;

    // workspace layout (2 MB total)
    _Float16* memF16 = (_Float16*)d_ws;                             // 512 KB
    _Float16* memT16 = memF16 + (size_t)N_HEADS * N_MEM * HEAD_DIM; // 512 KB
    float*    gate   = (float*)(memT16 + (size_t)N_HEADS * HEAD_DIM * N_MEM); // 1 MB

    PAR_memproj_kernel<<<1024, 256, 0, stream>>>(memory, Wm, bm, memF16, memT16);
    PAR_gate_kernel<<<(BATCH * T_TOK * N_HEADS) / 256, 256, 0, stream>>>(h, Wg, bg, gate);

    dim3 grid(T_TOK / T_TILE, N_HEADS, BATCH);   // (64, 16, 4)
    size_t shbytes = 106752;                      // 104.25 KB dynamic LDS
    PAR_attn_kernel<<<grid, 128, shbytes, stream>>>(h, memF16, memT16, gate, out);
}
